// EnhancedUVMDModel_56736517980848
// MI455X (gfx1250) — matrix operand order repaired and checked by reading, not compiled, not run on hardware
//
#include <hip/hip_runtime.h>

#define K_MODES 4
#define N_CH    12
#define FEAT    64
#define FFT_N   4096
#define FFT_LOGN 12
#define NSIG    768      // B*C = 64*12
#define NBINS   2049
#define BATCH   64
#define TWO_PI  6.28318530717958647692f
#define BN_EPS  1e-5f

typedef __attribute__((ext_vector_type(16))) _Float16 v16h;
typedef __attribute__((ext_vector_type(8)))  float    v8f;

// --------------------------------------------------------------------------
// Tensor Data Mover availability (5-arg on ROCm7.2/clang-22, 6-arg on clang-23)
// --------------------------------------------------------------------------
#if defined(__has_builtin)
#  if __has_builtin(__builtin_amdgcn_tensor_load_to_lds)
#    define HAVE_TDM 1
#  endif
#endif
#ifndef HAVE_TDM
#  define HAVE_TDM 0
#endif

#if HAVE_TDM
typedef __attribute__((ext_vector_type(4))) unsigned int u32x4;
typedef __attribute__((ext_vector_type(8))) int          i32x8;
typedef __attribute__((ext_vector_type(4))) int          i32x4;
#endif

// ---------------------------------------------------------------------------
// WMMA helpers (gfx1250: D(16x16,f32) = A(16x32,f16) x B(32x16,f16) + C)
// ---------------------------------------------------------------------------
__device__ __forceinline__ v8f wmma_f16(v16h a, v16h b, v8f c) {
    return __builtin_amdgcn_wmma_f32_16x16x32_f16(false, a, false, b,
                                                  (short)0, c, false, false);
}

// Pre-swizzled A fragment: one contiguous 32-byte vector per lane.
__device__ __forceinline__ v16h load_afrag(const _Float16* base, int fragIdx, int lane) {
    return *(const v16h*)(base + ((size_t)fragIdx * 32 + lane) * 16);
}

// B fragment (32x16 KxN, f16) gathered from an LDS [channel][column] tile.
// Layout (ISA 7.12.2): K = (lane/16)*8 + (h<8 ? h : h+8), N = lane%16.
// K decodes as (in_channel, tap): kk = i*KW + r ; column = colBase + N + r.
template <int KW, int KMAX>
__device__ __forceinline__ v16h load_bfrag(const _Float16* lds, int pitch,
                                           int kBase, int colBase, int lane) {
    v16h bf;
    const int n   = lane & 15;
    const int kHi = lane & 16;        // 0 or 16
    #pragma unroll
    for (int h = 0; h < 16; ++h) {
        int kk = kBase + (kHi >> 1) + (h < 8 ? h : h + 8);
        _Float16 v = (_Float16)0.f;
        if (kk < KMAX) {
            int i = kk / KW;
            int r = kk - i * KW;
            v = lds[i * pitch + colBase + n + r];
        }
        bf[h] = v;
    }
    return bf;
}

// Layer-1 variant: additionally masks columns whose ABSOLUTE time index falls
// outside [0, FFT_N) — implements 'same' zero padding and makes TDM edge
// over-reads harmless.
template <int KW, int KMAX>
__device__ __forceinline__ v16h load_bfrag_edge(const _Float16* lds, int pitch,
                                                int kBase, int colBase, int lane,
                                                int tAbs0) {
    v16h bf;
    const int n   = lane & 15;
    const int kHi = lane & 16;
    #pragma unroll
    for (int h = 0; h < 16; ++h) {
        int kk = kBase + (kHi >> 1) + (h < 8 ? h : h + 8);
        _Float16 v = (_Float16)0.f;
        if (kk < KMAX) {
            int i = kk / KW;
            int r = kk - i * KW;
            int col = colBase + n + r;
            if ((unsigned)(tAbs0 + col) < (unsigned)FFT_N)
                v = lds[i * pitch + col];
        }
        bf[h] = v;
    }
    return bf;
}

// C/D layout (ISA 7.12.2): M = v + 8*(lane/16), N = lane%16.
__device__ __forceinline__ void store_cfrag_relu(v8f c, _Float16* lds, int pitch,
                                                 int mBase, int colBase, int lane,
                                                 const float* sc, const float* bi) {
    const int n   = lane & 15;
    const int mHi = (lane & 16) >> 1;  // 0 or 8
    #pragma unroll
    for (int v = 0; v < 8; ++v) {
        int m = mBase + mHi + v;
        float y = fmaxf(c[v] * sc[m] + bi[m], 0.f);
        lds[m * pitch + colBase + n] = (_Float16)y;
    }
}

__device__ __forceinline__ void accum_cfrag_relu(v8f c, float* acc, int mBase, int lane,
                                                 const float* sc, const float* bi) {
    const int mHi = (lane & 16) >> 1;
    #pragma unroll
    for (int v = 0; v < 8; ++v) {
        int m = mBase + mHi + v;
        float y = fmaxf(c[v] * sc[m] + bi[m], 0.f);
        atomicAdd(&acc[m], y);   // ds_add_f32
    }
}

// ---------------------------------------------------------------------------
// 1) UVMD transfer functions H_k(f): run the (linear-in-X) recurrence with X=1
// ---------------------------------------------------------------------------
__global__ __launch_bounds__(256) void uvmd_H_kernel(const float* __restrict__ alpha, // [8][4]
                                                     const float* __restrict__ tau,   // [8]
                                                     const float* __restrict__ omega, // [4]
                                                     float* __restrict__ H) {         // [4][2049]
    int f = blockIdx.x * blockDim.x + threadIdx.x;
    if (f >= NBINS) return;
    float freq = 0.5f * (float)f / 2048.f;
    float u[K_MODES] = {0.f, 0.f, 0.f, 0.f};
    float total = 0.f, lam = 0.f;
    for (int l = 0; l < 8; ++l) {
        #pragma unroll
        for (int k = 0; k < K_MODES; ++k) {
            float resid = 1.f - (total - u[k]) + 0.5f * lam;
            float d = freq - omega[k];
            float denom = 1.f + alpha[l * K_MODES + k] * d * d;
            float nu = resid / denom;
            total += nu - u[k];
            u[k] = nu;
        }
        lam += tau[l] * (1.f - total);
    }
    #pragma unroll
    for (int k = 0; k < K_MODES; ++k) H[k * NBINS + f] = u[k];
}

// ---------------------------------------------------------------------------
// Shared radix-2 LDS FFT stages (input pre-bit-reversed), 256 threads/WG
// ---------------------------------------------------------------------------
__device__ __forceinline__ void fft_stages(float* re, float* im, int tid, float sign) {
    for (int stage = 1; stage <= FFT_LOGN; ++stage) {
        int half = 1 << (stage - 1);
        int m    = half << 1;
        for (int j = tid; j < FFT_N / 2; j += 256) {
            int grp  = j >> (stage - 1);
            int posn = j & (half - 1);
            int i1 = grp * m + posn;
            int i2 = i1 + half;
            float ang = sign * TWO_PI * (float)posn / (float)m;
            float wi, wr;
            __sincosf(ang, &wi, &wr);
            float br = re[i2], bi = im[i2];
            float tr = wr * br - wi * bi;
            float ti = wr * bi + wi * br;
            float ar = re[i1], ai = im[i1];
            re[i2] = ar - tr; im[i2] = ai - ti;
            re[i1] = ar + tr; im[i1] = ai + ti;
        }
        __syncthreads();
    }
}

// ---------------------------------------------------------------------------
// 2) Forward FFT: x (B,T,C) -> half spectrum [sig][0..2048] complex
// ---------------------------------------------------------------------------
__global__ __launch_bounds__(256) void fft_fwd_kernel(const float* __restrict__ x,
                                                      float2* __restrict__ spec) {
    __shared__ float re[FFT_N];
    __shared__ float im[FFT_N];
    int s = blockIdx.x;                 // signal = b*12 + c
    int b = s / N_CH, c = s % N_CH;
    const float* xp = x + ((size_t)b * FFT_N * N_CH + c);
    int tid = threadIdx.x;
    for (int t = tid; t < FFT_N; t += 256) {
        int rt = __brev((unsigned)t) >> (32 - FFT_LOGN);
        re[rt] = xp[(size_t)t * N_CH];
        im[rt] = 0.f;
    }
    __syncthreads();
    fft_stages(re, im, tid, -1.f);
    float2* sp = spec + (size_t)s * NBINS;
    for (int t = tid; t < NBINS; t += 256) {
        float2 v; v.x = re[t]; v.y = im[t];
        sp[t] = v;
    }
}

// ---------------------------------------------------------------------------
// 3) Inverse FFT per (mode, signal): apply H_k, iFFT, emit f16 mode signal
//    modes layout: [k][b][c][t]  == [(k*768 + s)][t]
// ---------------------------------------------------------------------------
__global__ __launch_bounds__(256) void ifft_modes_kernel(const float2* __restrict__ spec,
                                                         const float* __restrict__ H,
                                                         _Float16* __restrict__ modes) {
    __shared__ float re[FFT_N];
    __shared__ float im[FFT_N];
    int wg = blockIdx.x;
    int s  = wg >> 2;                  // signal
    int k  = wg & 3;                   // mode
    const float2* sp = spec + (size_t)s * NBINS;
    const float*  Hk = H + k * NBINS;
    int tid = threadIdx.x;
    for (int t = tid; t < FFT_N; t += 256) {
        int fold = (t <= 2048) ? t : (FFT_N - t);
        float2 v = sp[fold];
        float vy = (t <= 2048) ? v.y : -v.y;   // conjugate-symmetric extension
        float h = Hk[fold];
        int rt = __brev((unsigned)t) >> (32 - FFT_LOGN);
        re[rt] = v.x * h;
        im[rt] = vy * h;
    }
    __syncthreads();
    fft_stages(re, im, tid, +1.f);
    _Float16* mp = modes + ((size_t)k * NSIG + s) * FFT_N;
    const float inv = 1.f / (float)FFT_N;
    for (int t = tid; t < FFT_N; t += 256)
        mp[t] = (_Float16)(re[t] * inv);
}

// ---------------------------------------------------------------------------
// 4a) Weight pre-swizzle into WMMA A-fragment layout (f16), K padded to 32.
//     A-matrix 16x32 f16 layout (ISA 7.12.2):
//       K(lane,v,pos) = (v<4?0:16) + (v&3)*2 + pos + (lane/16)*8 ; M = lane%16
// ---------------------------------------------------------------------------
__global__ __launch_bounds__(256) void prep_wfrag_kernel(const float* __restrict__ W,
                                                         _Float16* __restrict__ out,
                                                         int O, int I, int kw,
                                                         int MT, int KT) {
    int idx = blockIdx.x * blockDim.x + threadIdx.x;
    int total = K_MODES * MT * KT * 32 * 16;
    if (idx >= total) return;
    int h    = idx & 15;
    int lane = (idx >> 4) & 31;
    int rest = idx >> 9;
    int kt = rest % KT; rest /= KT;
    int mt = rest % MT;
    int k  = rest / MT;
    int v = h >> 1, pos = h & 1;
    int klocal = ((v < 4) ? 0 : 16) + ((v & 3) << 1) + pos + ((lane >> 4) << 3);
    int kk = kt * 32 + klocal;
    int M  = mt * 16 + (lane & 15);
    float val = 0.f;
    if (kk < I * kw && M < O) {
        int i = kk / kw, r = kk - i * kw;
        val = W[(((size_t)k * O + M) * I + i) * kw + r];
    }
    out[idx] = (_Float16)val;
}

// 4b) Fold conv-bias + BN(eval) into per-channel scale/bias: y = s*conv + t
__global__ __launch_bounds__(256) void prep_sb_kernel(const float* __restrict__ g,
                                                      const float* __restrict__ bconv,
                                                      const float* __restrict__ be,
                                                      float* __restrict__ sb, int O) {
    int idx = blockIdx.x * blockDim.x + threadIdx.x;
    if (idx >= K_MODES * O) return;
    int k = idx / O, o = idx % O;
    float s = g[idx] * rsqrtf(1.f + BN_EPS);
    sb[(k * 2 + 0) * O + o] = s;
    sb[(k * 2 + 1) * O + o] = s * bconv[idx] + be[idx];
}

// ---------------------------------------------------------------------------
// 5) Fused 3-layer conv encoder + mean pool, one WG per (k,b), WMMA f16.
//    T processed in 16 chunks of 256 columns; input tile staged by the
//    Tensor Data Mover (double-buffered, overlapped with compute).
// ---------------------------------------------------------------------------
#define TC     256
#define NCHUNK (FFT_N / TC)
#define IN_W   288   // 12 rows : columns t0-6 .. t0+281 (edge-masked in bfrag)
#define H1_W   280   // 32 rows : 17 N-tiles (272 cols) + 8 zero cols
#define H2_W   272   // 64 rows : 17 N-tiles exactly

#if HAVE_TDM
// Issue one 2D TDM tile load: 12 rows x IN_W f16 cols, row stride FFT_N elems.
// D# per ISA ch.8: group0 = {count/type/lds_addr/global_addr},
// group1 = {data_size, tensor dims, tile dims, strides}. 2D -> groups 2/3 zero.
__device__ __forceinline__ void tdm_issue_tile(const _Float16* gsrc, unsigned ldsByteOff) {
    unsigned long long ga = (unsigned long long)(uintptr_t)gsrc;
    u32x4 g0;
    g0[0] = 1u;                                              // count=1, user mode
    g0[1] = ldsByteOff;                                      // lds_addr (bytes)
    g0[2] = (unsigned)ga;                                    // global_addr[31:0]
    g0[3] = (unsigned)((ga >> 32) & 0x01ffffffu) | (2u << 30); // ga[56:32] | type=2
    i32x8 g1;
    const unsigned TD0 = 1u << 20;                           // large: no OOB clip (edges masked in compute)
    g1[0] = (int)(1u << 16);                                 // data_size=1 -> 2 bytes
    g1[1] = (int)((TD0 & 0xffffu) << 16);                    // tensor_dim0[15:0]
    g1[2] = (int)((TD0 >> 16) | (16u << 16));                // tensor_dim0[31:16] | tensor_dim1=16
    g1[3] = (int)((unsigned)IN_W << 16);                     // tile_dim0 = 288
    g1[4] = N_CH;                                            // tile_dim1 = 12, tile_dim2 = 0
    g1[5] = FFT_N;                                           // tensor_dim0_stride = 4096
    g1[6] = 0;
    g1[7] = 0;
    i32x4 z4 = {0, 0, 0, 0};
#if __clang_major__ >= 23
    i32x8 z8 = {0, 0, 0, 0, 0, 0, 0, 0};
    __builtin_amdgcn_tensor_load_to_lds(g0, g1, z4, z4, z8, 0);
#else
    __builtin_amdgcn_tensor_load_to_lds(g0, g1, z4, z4, 0);
#endif
}
#endif

__global__ __launch_bounds__(256, 1) void encoder_kernel(
    const _Float16* __restrict__ modes,                       // [4][768][4096]
    const _Float16* __restrict__ A1,                          // [4][2*3][32][16]
    const _Float16* __restrict__ A2,                          // [4][4*5][32][16]
    const _Float16* __restrict__ A3,                          // [4][4*6][32][16]
    const float* __restrict__ sb1, const float* __restrict__ sb2,
    const float* __restrict__ sb3,
    float* __restrict__ feats)                                // [64][256]
{
    __shared__ _Float16 s_in[2][N_CH * IN_W];
    __shared__ _Float16 s_h1[32 * H1_W];
    __shared__ _Float16 s_h2[64 * H2_W];
    __shared__ float s_acc[FEAT];
    __shared__ float s_s1[32], s_b1[32], s_s2[64], s_b2[64], s_s3[64], s_b3[64];

    const int wg   = blockIdx.x;         // k*64 + b
    const int k    = wg >> 6;
    const int b    = wg & 63;
    const int tid  = threadIdx.x;
    const int lane = tid & 31;
    const int wave = tid >> 5;

    if (tid < 32)       { s_s1[tid] = sb1[(k*2+0)*32 + tid];   s_b1[tid] = sb1[(k*2+1)*32 + tid]; }
    else if (tid < 96)  { int j = tid-32; s_s2[j] = sb2[(k*2+0)*64 + j]; s_b2[j] = sb2[(k*2+1)*64 + j]; }
    else if (tid < 160) { int j = tid-96; s_s3[j] = sb3[(k*2+0)*64 + j]; s_b3[j] = sb3[(k*2+1)*64 + j]; }
    for (int i = tid; i < 32 * H1_W; i += 256) s_h1[i] = (_Float16)0.f;  // pad cols stay zero
    if (tid < FEAT) s_acc[tid] = 0.f;

    const _Float16* mp  = modes + (size_t)wg * N_CH * FFT_N;
    const _Float16* A1k = A1 + (size_t)k * (2 * 3 * 512);
    const _Float16* A2k = A2 + (size_t)k * (4 * 5 * 512);
    const _Float16* A3k = A3 + (size_t)k * (4 * 6 * 512);

#if HAVE_TDM
    if (wave == 0)
        tdm_issue_tile(mp - 6, (unsigned)(uintptr_t)(void*)&s_in[0][0]);
#endif
    __syncthreads();

    for (int chunk = 0; chunk < NCHUNK; ++chunk) {
        const int t0  = chunk * TC;
        const int buf = chunk & 1;

#if HAVE_TDM
        if (wave == 0) __builtin_amdgcn_s_wait_tensorcnt(0);
        __syncthreads();                    // current tile visible to all waves
        if (wave == 0 && chunk + 1 < NCHUNK)
            tdm_issue_tile(mp + (t0 + TC - 6),
                           (unsigned)(uintptr_t)(void*)&s_in[buf ^ 1][0]);
#else
        for (int idx = tid; idx < N_CH * IN_W; idx += 256) {
            int ch = idx / IN_W, ci = idx - ch * IN_W;
            int t = t0 - 6 + ci;
            _Float16 v = (_Float16)0.f;
            if (t >= 0 && t < FFT_N) v = mp[ch * FFT_N + t];
            s_in[buf][ch * IN_W + ci] = v;
        }
        __syncthreads();
#endif

        // ---- conv1: O=32 (2 mtiles), K=12*7=84 (3 ktiles), kw=7 ----
        for (int nt = wave; nt < 17; nt += 8) {
            v8f c0 = {}, c1 = {};
            #pragma unroll
            for (int kt = 0; kt < 3; ++kt) {
                v16h bf = load_bfrag_edge<7, 84>(&s_in[buf][0], IN_W, kt * 32,
                                                 nt * 16, lane, t0 - 6);
                c0 = wmma_f16(load_afrag(A1k, 0 * 3 + kt, lane), bf, c0);
                c1 = wmma_f16(load_afrag(A1k, 1 * 3 + kt, lane), bf, c1);
            }
            store_cfrag_relu(c0, s_h1, H1_W,  0, nt * 16, lane, s_s1, s_b1);
            store_cfrag_relu(c1, s_h1, H1_W, 16, nt * 16, lane, s_s1, s_b1);
        }
        __syncthreads();

        // ---- conv2: O=64 (4 mtiles), K=32*5=160 (5 ktiles), kw=5 ----
        for (int nt = wave; nt < 17; nt += 8) {
            v8f c0 = {}, c1 = {}, c2 = {}, c3 = {};
            #pragma unroll
            for (int kt = 0; kt < 5; ++kt) {
                v16h bf = load_bfrag<5, 160>(s_h1, H1_W, kt * 32, nt * 16, lane);
                c0 = wmma_f16(load_afrag(A2k, 0 * 5 + kt, lane), bf, c0);
                c1 = wmma_f16(load_afrag(A2k, 1 * 5 + kt, lane), bf, c1);
                c2 = wmma_f16(load_afrag(A2k, 2 * 5 + kt, lane), bf, c2);
                c3 = wmma_f16(load_afrag(A2k, 3 * 5 + kt, lane), bf, c3);
            }
            store_cfrag_relu(c0, s_h2, H2_W,  0, nt * 16, lane, s_s2, s_b2);
            store_cfrag_relu(c1, s_h2, H2_W, 16, nt * 16, lane, s_s2, s_b2);
            store_cfrag_relu(c2, s_h2, H2_W, 32, nt * 16, lane, s_s2, s_b2);
            store_cfrag_relu(c3, s_h2, H2_W, 48, nt * 16, lane, s_s2, s_b2);
        }
        __syncthreads();

        // ---- conv3: O=64 (4 mtiles), K=64*3=192 (6 ktiles), kw=3 + mean ----
        for (int nt = wave; nt < 16; nt += 8) {
            v8f c0 = {}, c1 = {}, c2 = {}, c3 = {};
            #pragma unroll
            for (int kt = 0; kt < 6; ++kt) {
                v16h bf = load_bfrag<3, 192>(s_h2, H2_W, kt * 32, nt * 16, lane);
                c0 = wmma_f16(load_afrag(A3k, 0 * 6 + kt, lane), bf, c0);
                c1 = wmma_f16(load_afrag(A3k, 1 * 6 + kt, lane), bf, c1);
                c2 = wmma_f16(load_afrag(A3k, 2 * 6 + kt, lane), bf, c2);
                c3 = wmma_f16(load_afrag(A3k, 3 * 6 + kt, lane), bf, c3);
            }
            accum_cfrag_relu(c0, s_acc,  0, lane, s_s3, s_b3);
            accum_cfrag_relu(c1, s_acc, 16, lane, s_s3, s_b3);
            accum_cfrag_relu(c2, s_acc, 32, lane, s_s3, s_b3);
            accum_cfrag_relu(c3, s_acc, 48, lane, s_s3, s_b3);
        }
        __syncthreads();
    }

    if (tid < FEAT)
        feats[b * (K_MODES * FEAT) + k * FEAT + tid] = s_acc[tid] * (1.f / (float)FFT_N);
}

// ---------------------------------------------------------------------------
// 6) Classifier: relu(feat @ Wc1^T + bc1) @ Wc2^T + bc2   (tiny; VALU)
// ---------------------------------------------------------------------------
__global__ __launch_bounds__(128) void classifier_kernel(const float* __restrict__ feats,
                                                         const float* __restrict__ Wc1,
                                                         const float* __restrict__ bc1,
                                                         const float* __restrict__ Wc2,
                                                         const float* __restrict__ bc2,
                                                         float* __restrict__ out) {
    __shared__ float s_f[256];
    __shared__ float s_h[128];
    int b = blockIdx.x, tid = threadIdx.x;
    for (int i = tid; i < 256; i += 128) s_f[i] = feats[b * 256 + i];
    __syncthreads();
    float acc = bc1[tid];
    for (int i = 0; i < 256; ++i) acc += s_f[i] * Wc1[tid * 256 + i];
    s_h[tid] = fmaxf(acc, 0.f);
    __syncthreads();
    if (tid < 10) {
        float o = bc2[tid];
        for (int j = 0; j < 128; ++j) o += s_h[j] * Wc2[tid * 128 + j];
        out[b * 10 + tid] = o;
    }
}

// ---------------------------------------------------------------------------
// Host-side launch
// ---------------------------------------------------------------------------
extern "C" void kernel_launch(void* const* d_in, const int* in_sizes, int n_in,
                              void* d_out, int out_size, void* d_ws, size_t ws_size,
                              hipStream_t stream) {
    const float* x     = (const float*)d_in[0];
    const float* alpha = (const float*)d_in[1];
    const float* tau   = (const float*)d_in[2];
    const float* omega = (const float*)d_in[3];
    const float* W1    = (const float*)d_in[4];
    const float* b1    = (const float*)d_in[5];
    const float* g1    = (const float*)d_in[6];
    const float* be1   = (const float*)d_in[7];
    const float* W2    = (const float*)d_in[8];
    const float* b2    = (const float*)d_in[9];
    const float* g2    = (const float*)d_in[10];
    const float* be2   = (const float*)d_in[11];
    const float* W3    = (const float*)d_in[12];
    const float* b3    = (const float*)d_in[13];
    const float* g3    = (const float*)d_in[14];
    const float* be3   = (const float*)d_in[15];
    const float* Wc1   = (const float*)d_in[16];
    const float* bc1   = (const float*)d_in[17];
    const float* Wc2   = (const float*)d_in[18];
    const float* bc2   = (const float*)d_in[19];
    float* out = (float*)d_out;

    char*  ws  = (char*)d_ws;
    size_t off = 0;
    auto alloc = [&](size_t bytes) -> void* {
        void* p = ws + off;
        off = (off + bytes + 255) & ~(size_t)255;
        return p;
    };
    float*    H     = (float*)   alloc((size_t)K_MODES * NBINS * 4);
    float2*   spec  = (float2*)  alloc((size_t)NSIG * NBINS * 8);
    _Float16* modes = (_Float16*)alloc((size_t)K_MODES * NSIG * FFT_N * 2);
    _Float16* A1    = (_Float16*)alloc((size_t)K_MODES * 2 * 3 * 512 * 2);
    _Float16* A2    = (_Float16*)alloc((size_t)K_MODES * 4 * 5 * 512 * 2);
    _Float16* A3    = (_Float16*)alloc((size_t)K_MODES * 4 * 6 * 512 * 2);
    float*    sb1   = (float*)   alloc((size_t)K_MODES * 2 * 32 * 4);
    float*    sb2   = (float*)   alloc((size_t)K_MODES * 2 * 64 * 4);
    float*    sb3   = (float*)   alloc((size_t)K_MODES * 2 * 64 * 4);
    float*    feats = (float*)   alloc((size_t)BATCH * K_MODES * FEAT * 4);

    // weight prep
    prep_sb_kernel<<<1, 128, 0, stream>>>(g1, b1, be1, sb1, 32);
    prep_sb_kernel<<<1, 256, 0, stream>>>(g2, b2, be2, sb2, 64);
    prep_sb_kernel<<<1, 256, 0, stream>>>(g3, b3, be3, sb3, 64);
    prep_wfrag_kernel<<<(K_MODES*2*3*512 + 255)/256, 256, 0, stream>>>(W1, A1, 32, 12, 7, 2, 3);
    prep_wfrag_kernel<<<(K_MODES*4*5*512 + 255)/256, 256, 0, stream>>>(W2, A2, 64, 32, 5, 4, 5);
    prep_wfrag_kernel<<<(K_MODES*4*6*512 + 255)/256, 256, 0, stream>>>(W3, A3, 64, 64, 3, 4, 6);

    // UVMD as 4 real spectral filters
    uvmd_H_kernel<<<(NBINS + 255)/256, 256, 0, stream>>>(alpha, tau, omega, H);
    fft_fwd_kernel<<<NSIG, 256, 0, stream>>>(x, spec);
    ifft_modes_kernel<<<NSIG * K_MODES, 256, 0, stream>>>(spec, H, modes);

    // fused WMMA conv encoder + mean pool (TDM-staged, double buffered)
    encoder_kernel<<<K_MODES * BATCH, 256, 0, stream>>>(modes, A1, A2, A3, sb1, sb2, sb3, feats);

    // classifier head
    classifier_kernel<<<BATCH, 128, 0, stream>>>(feats, Wc1, bc1, Wc2, bc2, out);
}
